// SpatialMemoryBank_13486197309662
// MI455X (gfx1250) — matrix operand order repaired and checked by reading, not compiled, not run on hardware
//
#include <hip/hip_runtime.h>
#include <math.h>

#define B_ 256
#define D_ 384
#define H_ 512
#define M_ 200000
#define K_ 5
#define CHUNK_ 512
#define NCHUNK_ ((M_ + CHUNK_ - 1) / CHUNK_)   // 391
#define EPS_ 1e-8f

typedef __bf16 bf16;
typedef __attribute__((ext_vector_type(16))) __bf16 v16bf;
typedef __attribute__((ext_vector_type(8)))  __bf16 v8bf;
typedef __attribute__((ext_vector_type(8)))  float  v8f;
typedef __attribute__((ext_vector_type(2)))  float  v2f;

// ---------------- spatial activations: 1/(1+||coords - mean(sw)||) ----------------
__global__ void act_kernel(const float* __restrict__ sw, const float* __restrict__ coords,
                           float* __restrict__ act) {
  __shared__ float rx[256], ry[256];
  int t = threadIdx.x;
  rx[t] = sw[2*t]     + sw[2*(t+256)];
  ry[t] = sw[2*t + 1] + sw[2*(t+256) + 1];
  __syncthreads();
  for (int s = 128; s > 0; s >>= 1) {
    if (t < s) { rx[t] += rx[t+s]; ry[t] += ry[t+s]; }
    __syncthreads();
  }
  float cx = rx[0] * (1.0f/512.0f), cy = ry[0] * (1.0f/512.0f);
  int m = blockIdx.x * 256 + t;
  if (m < M_) {
    float dx = coords[2*m] - cx, dy = coords[2*m+1] - cy;
    act[m] = 1.0f / (1.0f + sqrtf(dx*dx + dy*dy));
  }
}

// ---------------- normalized bf16 queries ----------------
__global__ void qnorm_kernel(const float* __restrict__ q, bf16* __restrict__ qn) {
  __shared__ float red[128];
  int b = blockIdx.x, t = threadIdx.x;
  float s = 0.f;
  for (int i = t; i < D_; i += 128) { float v = q[b*D_ + i]; s += v*v; }
  red[t] = s; __syncthreads();
  for (int st = 64; st > 0; st >>= 1) { if (t < st) red[t] += red[t+st]; __syncthreads(); }
  float inv = 1.0f / fmaxf(sqrtf(red[0]), EPS_);
  for (int i = t; i < D_; i += 128) qn[b*D_ + i] = (bf16)(q[b*D_ + i] * inv);
}

// ---------------- fused: normalize mem rows -> bf16 WMMA sim -> per-chunk top-5 ----------------
// block: 512 threads = 16 waves; wave w owns query tile w (16 queries), chunk = 512 mem rows.
__global__ __launch_bounds__(512) void sim_topk_kernel(
    const bf16* __restrict__ qn, const float* __restrict__ mem,
    const float* __restrict__ act, float* __restrict__ pval, int* __restrict__ pidx)
{
  __shared__ bf16  bt[16][392];           // normalized bf16 mem tile, padded rows
  __shared__ float simBuf[16][16][17];    // per-wave 16x16 C tile staging
  __shared__ float topv[B_][K_];
  __shared__ int   topi[B_][K_];
  __shared__ float actT[16];

  const int tid  = threadIdx.x;
  const int lane = tid & 31, wave = tid >> 5;
  const int half = lane >> 4, l15 = lane & 15;

  // Preload this wave's A fragments (queries wave*16 .. wave*16+15, full K=384) into VGPRs.
  // 16-bit A 16x32 layout: lanes 0-15 hold K 0-7 (v0-3) & 16-23 (v4-7); lanes 16-31 hold K 8-15 & 24-31.
  v16bf afrag[12];
  {
    const bf16* row = qn + (size_t)(wave*16 + l15) * D_;
    #pragma unroll
    for (int ksi = 0; ksi < 12; ++ksi) {
      int ks = ksi * 32;
      v8bf lo = *(const v8bf*)(row + ks + half*8);
      v8bf hi = *(const v8bf*)(row + ks + 16 + half*8);
      afrag[ksi] = __builtin_shufflevector(lo, hi, 0,1,2,3,4,5,6,7,8,9,10,11,12,13,14,15);
    }
  }
  for (int i = tid; i < B_*K_; i += 512) { (&topv[0][0])[i] = -3.0e38f; (&topi[0][0])[i] = 0; }
  __syncthreads();

  const int base = blockIdx.x * CHUNK_;
  const int r = wave, seg = lane;         // 16 rows x 32 threads/row, 12 floats each

  for (int mt = 0; mt < CHUNK_/16; ++mt) {
    const int m0 = base + mt*16;
    __syncthreads();                      // previous tile fully consumed before overwrite
    if (tid < 16) { int gg = m0 + tid; actT[tid] = (gg < M_) ? act[gg] : -3.0e38f; }

    const int g = m0 + r;                 // uniform per wave
    float4 v4[3];
    if (g < M_) {
      const float4* p = (const float4*)(mem + (size_t)g * D_ + seg*12);
      v4[0] = p[0]; v4[1] = p[1]; v4[2] = p[2];
    } else {
      v4[0] = v4[1] = v4[2] = make_float4(0.f, 0.f, 0.f, 0.f);
    }
    float ss = v4[0].x*v4[0].x + v4[0].y*v4[0].y + v4[0].z*v4[0].z + v4[0].w*v4[0].w
             + v4[1].x*v4[1].x + v4[1].y*v4[1].y + v4[1].z*v4[1].z + v4[1].w*v4[1].w
             + v4[2].x*v4[2].x + v4[2].y*v4[2].y + v4[2].z*v4[2].z + v4[2].w*v4[2].w;
    #pragma unroll
    for (int off = 16; off > 0; off >>= 1) ss += __shfl_xor(ss, off, 32);  // row == wave
    float inv = 1.0f / fmaxf(sqrtf(ss), EPS_);
    bf16* brow = &bt[r][seg*12];
    #pragma unroll
    for (int i = 0; i < 3; ++i) {
      brow[4*i+0] = (bf16)(v4[i].x * inv);
      brow[4*i+1] = (bf16)(v4[i].y * inv);
      brow[4*i+2] = (bf16)(v4[i].z * inv);
      brow[4*i+3] = (bf16)(v4[i].w * inv);
    }
    __syncthreads();

    // WMMA: [16q x 384] x [384 x 16m].  B 32x16 layout: lane n (0-15) K 0-7 & 16-23, lane n+16 K 8-15 & 24-31.
    // Load ALL B fragments for this tile into registers first (24 x ds_load_b128 in flight),
    // then stream the 12 WMMAs over two independent accumulator chains. In-order LDS returns
    // let the compiler use partial s_wait_dscnt while later loads are still outstanding.
    const bf16* bbase = &bt[l15][0];
    v8bf blo[12], bhi[12];
    #pragma unroll
    for (int ksi = 0; ksi < 12; ++ksi) {
      blo[ksi] = *(const v8bf*)(bbase + ksi*32 + half*8);
      bhi[ksi] = *(const v8bf*)(bbase + ksi*32 + 16 + half*8);
    }
    v8f acc0 = {}, acc1 = {};
    #pragma unroll
    for (int ksi = 0; ksi < 12; ksi += 2) {
      v16bf b0 = __builtin_shufflevector(blo[ksi],   bhi[ksi],   0,1,2,3,4,5,6,7,8,9,10,11,12,13,14,15);
      v16bf b1 = __builtin_shufflevector(blo[ksi+1], bhi[ksi+1], 0,1,2,3,4,5,6,7,8,9,10,11,12,13,14,15);
      acc0 = __builtin_amdgcn_wmma_f32_16x16x32_bf16(false, afrag[ksi],   false, b0,
                                                     (short)0, acc0, false, false);
      acc1 = __builtin_amdgcn_wmma_f32_16x16x32_bf16(false, afrag[ksi+1], false, b1,
                                                     (short)0, acc1, false, false);
    }
    // C layout: vgpr i, lanes0-15 -> row i, lanes16-31 -> row i+8; col = l15
    #pragma unroll
    for (int i = 0; i < 8; ++i) simBuf[wave][i + 8*half][l15] = acc0[i] + acc1[i];
    if (lane < 16) {                      // lane owns one query row (wave-internal LDS, in-order)
      float* tv = topv[wave*16 + lane];
      int*   ti = topi[wave*16 + lane];
      for (int nl = 0; nl < 16; ++nl) {
        float val = simBuf[wave][lane][nl] + actT[nl];
        if (val > tv[K_-1]) {
          int idx = m0 + nl;
          int p = K_-1;
          while (p > 0 && val > tv[p-1]) { tv[p] = tv[p-1]; ti[p] = ti[p-1]; --p; }
          tv[p] = val; ti[p] = idx;
        }
      }
    }
  }
  __syncthreads();
  const size_t ob = (size_t)blockIdx.x * B_ * K_;
  for (int i = tid; i < B_*K_; i += 512) {
    pval[ob + i] = (&topv[0][0])[i];
    pidx[ob + i] = (&topi[0][0])[i];
  }
}

// ---------------- merge per-chunk partials -> global top-5 indices ----------------
__global__ void merge_kernel(const float* __restrict__ pval, const int* __restrict__ pidx,
                             int* __restrict__ tidx) {
  int b = threadIdx.x;
  float tv[K_]; int ti[K_];
  #pragma unroll
  for (int j = 0; j < K_; ++j) { tv[j] = -3.0e38f; ti[j] = 0; }
  for (int c = 0; c < NCHUNK_; ++c) {
    const float* pv = pval + ((size_t)c * B_ + b) * K_;
    const int*   pi = pidx + ((size_t)c * B_ + b) * K_;
    #pragma unroll
    for (int j = 0; j < K_; ++j) {
      float v = pv[j];
      if (v > tv[K_-1]) {
        int id = pi[j];
        int p = K_-1;
        while (p > 0 && v > tv[p-1]) { tv[p] = tv[p-1]; ti[p] = ti[p-1]; --p; }
        tv[p] = v; ti[p] = id;
      }
    }
  }
  #pragma unroll
  for (int j = 0; j < K_; ++j) tidx[b*K_ + j] = ti[j];
}

// ---------------- build [query, retrieved_1..5] rows ----------------
__global__ void gather_kernel(const float* __restrict__ q, const float* __restrict__ mem,
                              const int* __restrict__ tidx, float* __restrict__ cat) {
  int i = blockIdx.x * 256 + threadIdx.x;
  if (i >= B_*6*D_) return;
  int c = i % D_, rrow = i / D_;
  int b = rrow / 6, t = rrow % 6;
  cat[i] = (t == 0) ? q[b*D_ + c] : mem[(size_t)tidx[b*K_ + t - 1] * D_ + c];
}

// ---------------- generic Y = act(X @ W^T + bias [+ addend]) using f32 WMMA ----------------
__global__ __launch_bounds__(256) void gemm16_kernel(
    const float* __restrict__ X, const float* __restrict__ W, const float* __restrict__ bias,
    const float* __restrict__ addend, int addStride, float* __restrict__ Y,
    int R, int N, int Dk, int actMode)
{
  int lane = threadIdx.x & 31, wave = threadIdx.x >> 5;
  int ntN = N >> 4;
  int tile = blockIdx.x * 8 + wave;
  if (tile >= (R >> 4) * ntN) return;     // wave-uniform
  int rT = tile / ntN, cT = tile - rT * ntN;
  int half = lane >> 4, l15 = lane & 15;
  int m = rT*16 + l15, n = cT*16 + l15;
  // f32 A 16x4: lanes 0-15 K0(v0),K1(v1); lanes 16-31 K2,K3.  B 4x16 mirrored.
  const float* xrow = X + (size_t)m * Dk + 2*half;
  const float* wrow = W + (size_t)n * Dk + 2*half;
  v8f acc = {};
  #pragma unroll 4
  for (int k = 0; k < Dk; k += 4) {
    v2f a = *(const v2f*)(xrow + k);
    v2f b = *(const v2f*)(wrow + k);
    acc = __builtin_amdgcn_wmma_f32_16x16x4_f32(false, a, false, b, (short)0, acc, false, false);
  }
  float bs = bias[n];
  #pragma unroll
  for (int i = 0; i < 8; ++i) {
    int row = rT*16 + i + 8*half;
    float v = acc[i] + bs;
    if (addend) v += addend[(size_t)row * addStride + n];
    if (actMode == 1) v = 1.0f / (1.0f + expf(-v));
    else if (actMode == 2) v = tanhf(v);
    Y[(size_t)row * N + n] = v;
  }
}

__global__ void zero_kernel(float* p, int n) {
  int i = blockIdx.x * 256 + threadIdx.x;
  if (i < n) p[i] = 0.f;
}

__global__ void combine_kernel(const float* __restrict__ g, const float* __restrict__ h,
                               const float* __restrict__ xp, float* __restrict__ out) {
  int i = blockIdx.x * 256 + threadIdx.x;
  if (i >= B_*H_) return;
  int b = i / H_, j = i - b*H_;
  float gv = g[i];
  out[i] = gv * h[i] + (1.0f - gv) * xp[(size_t)b * 6 * H_ + j];   // xp[:,0,:] = projected_query
}

extern "C" void kernel_launch(void* const* d_in, const int* in_sizes, int n_in,
                              void* d_out, int out_size, void* d_ws, size_t ws_size,
                              hipStream_t stream) {
  (void)in_sizes; (void)n_in; (void)out_size; (void)ws_size;
  const float* query  = (const float*)d_in[0];
  const float* memv   = (const float*)d_in[1];
  const float* coords = (const float*)d_in[2];
  const float* sw     = (const float*)d_in[3];
  const float* W_ih   = (const float*)d_in[4];
  const float* b_ih   = (const float*)d_in[5];
  const float* W_hh   = (const float*)d_in[6];
  const float* b_hh   = (const float*)d_in[7];
  const float* gate_W = (const float*)d_in[8];
  const float* gate_b = (const float*)d_in[9];
  float* out = (float*)d_out;

  char* ws = (char*)d_ws;
  size_t off = 0;
  auto alloc = [&](size_t bytes) -> void* {
    void* p = ws + off; off += (bytes + 255) & ~(size_t)255; return p;
  };
  float* act   = (float*)alloc((size_t)M_ * 4);
  bf16*  qn    = (bf16*) alloc((size_t)B_ * D_ * 2);
  float* pval  = (float*)alloc((size_t)NCHUNK_ * B_ * K_ * 4);
  int*   pidx  = (int*)  alloc((size_t)NCHUNK_ * B_ * K_ * 4);
  int*   tidx  = (int*)  alloc((size_t)B_ * K_ * 4);
  float* cat   = (float*)alloc((size_t)B_ * 6 * D_ * 4);
  float* xproj = (float*)alloc((size_t)B_ * 6 * H_ * 4);
  float* gatep = (float*)alloc((size_t)B_ * H_ * 4);
  float* h0    = (float*)alloc((size_t)B_ * H_ * 4);
  float* h1    = (float*)alloc((size_t)B_ * H_ * 4);

  act_kernel<<<(M_ + 255)/256, 256, 0, stream>>>(sw, coords, act);
  qnorm_kernel<<<B_, 128, 0, stream>>>(query, qn);
  sim_topk_kernel<<<NCHUNK_, 512, 0, stream>>>(qn, memv, act, pval, pidx);
  merge_kernel<<<1, B_, 0, stream>>>(pval, pidx, tidx);
  gather_kernel<<<(B_*6*D_ + 255)/256, 256, 0, stream>>>(query, memv, tidx, cat);

  // x_proj for all 6 timesteps: [1536 x 384] @ W_ih^T -> [1536 x 512]
  gemm16_kernel<<<((B_*6/16)*(H_/16))/8, 256, 0, stream>>>(cat, W_ih, b_ih, nullptr, 0,
                                                           xproj, B_*6, H_, D_, 0);
  // gate = sigmoid(query @ gate_W^T + gate_b)
  gemm16_kernel<<<((B_/16)*(H_/16))/8, 256, 0, stream>>>(query, gate_W, gate_b, nullptr, 0,
                                                         gatep, B_, H_, D_, 1);
  zero_kernel<<<(B_*H_ + 255)/256, 256, 0, stream>>>(h0, B_*H_);
  float* hin = h0; float* hout = h1;
  for (int t = 0; t < 6; ++t) {
    gemm16_kernel<<<((B_/16)*(H_/16))/8, 256, 0, stream>>>(hin, W_hh, b_hh,
                                                           xproj + t*H_, 6*H_,
                                                           hout, B_, H_, H_, 2);
    float* tmp = hin; hin = hout; hout = tmp;
  }
  combine_kernel<<<(B_*H_ + 255)/256, 256, 0, stream>>>(gatep, hin, xproj, out);
}
